// ConMambaCTC_62062277427576
// MI455X (gfx1250) — compile-verified, matrix operand order checked by reading, not run on hardware
//
#include <hip/hip_runtime.h>
#include <hip/hip_bf16.h>
#include <math.h>

// ---------------------------------------------------------------------------
// ConMamba CTC for MI455X (gfx1250). GEMMs via v_wmma_f32_16x16x32_f16
// (f16 in, f32 accumulate), register-blocked 64x32 per wave.
// Scan / norms / activations in f32 VALU. conv2 lowered to im2col + WMMA.
// ---------------------------------------------------------------------------

typedef __attribute__((ext_vector_type(16))) _Float16 v16h;
typedef __attribute__((ext_vector_type(8)))  _Float16 v8h;
typedef __attribute__((ext_vector_type(8)))  float    v8f;

__device__ __forceinline__ float sigmoidf_(float x){ return 1.f/(1.f+__expf(-x)); }
__device__ __forceinline__ float siluf_(float x){ return x*sigmoidf_(x); }
__device__ __forceinline__ float geluf_(float x){ return 0.5f*x*(1.f+erff(x*0.70710678118654752f)); }

// ------------------------------ WMMA GEMM ----------------------------------
// D[M,N] = A[M,K](f16,row-major,lda) * Bt[N,K](f16,row-major,ldb)^T
//          (+ Cin[M,N]) (+ bias[N]) (act==1 -> GELU epilogue)
// block = 128 threads (4 waves, 2x2); wave -> 64x32 tile (4 M x 2 N subtiles)
// grid = (N/64, M/128); M % 128 == 0, N % 64 == 0, K % 32 == 0.
__global__ void __launch_bounds__(128)
k_wmma_gemm(const _Float16* __restrict__ A, int lda,
            const _Float16* __restrict__ Bt, int ldb,
            const float* __restrict__ Cin, const float* __restrict__ bias,
            float* __restrict__ D, int ldd, int K, int act)
{
    const int lane = threadIdx.x & 31;
    const int wave = threadIdx.x >> 5;          // 0..3
    const int wm = wave >> 1, wn = wave & 1;
    const int rowW = blockIdx.y * 128 + wm * 64;   // wave M origin
    const int colW = blockIdx.x * 64  + wn * 32;   // wave N origin
    const int rsel = lane & 15;
    const int hi   = lane >> 4;
    const int koff = hi * 8;                       // per documented f16 layout

    const _Float16* ap[4];
    const _Float16* bp[2];
#pragma unroll
    for (int m = 0; m < 4; ++m) ap[m] = A  + (size_t)(rowW + 16*m + rsel) * lda + koff;
#pragma unroll
    for (int n = 0; n < 2; ++n) bp[n] = Bt + (size_t)(colW + 16*n + rsel) * ldb + koff;

    v8f acc[4][2];
    if (Cin) {
#pragma unroll
        for (int m = 0; m < 4; ++m)
#pragma unroll
            for (int n = 0; n < 2; ++n)
#pragma unroll
                for (int r = 0; r < 8; ++r)
                    acc[m][n][r] = Cin[(size_t)(rowW + 16*m + r + 8*hi) * ldd
                                       + colW + 16*n + rsel];
    } else {
#pragma unroll
        for (int m = 0; m < 4; ++m)
#pragma unroll
            for (int n = 0; n < 2; ++n)
#pragma unroll
                for (int r = 0; r < 8; ++r) acc[m][n][r] = 0.f;
    }

    for (int kb = 0; kb < K; kb += 32) {
        v16h af[4], bf[2];
#pragma unroll
        for (int m = 0; m < 4; ++m) {
            __builtin_prefetch(ap[m] + kb + 32, 0, 1);     // global_prefetch_b8
            v8h lo  = *(const v8h*)(ap[m] + kb);
            v8h hi8 = *(const v8h*)(ap[m] + kb + 16);
#pragma unroll
            for (int e = 0; e < 8; ++e) { af[m][e] = lo[e]; af[m][e+8] = hi8[e]; }
        }
#pragma unroll
        for (int n = 0; n < 2; ++n) {
            __builtin_prefetch(bp[n] + kb + 32, 0, 1);
            v8h lo  = *(const v8h*)(bp[n] + kb);
            v8h hi8 = *(const v8h*)(bp[n] + kb + 16);
#pragma unroll
            for (int e = 0; e < 8; ++e) { bf[n][e] = lo[e]; bf[n][e+8] = hi8[e]; }
        }
#pragma unroll
        for (int m = 0; m < 4; ++m)
#pragma unroll
            for (int n = 0; n < 2; ++n)
                acc[m][n] = __builtin_amdgcn_wmma_f32_16x16x32_f16(
                                false, af[m], false, bf[n], (short)0,
                                acc[m][n], false, false);
    }

    float bv[2] = {0.f, 0.f};
    if (bias) {
#pragma unroll
        for (int n = 0; n < 2; ++n) bv[n] = bias[colW + 16*n + rsel];
    }
#pragma unroll
    for (int m = 0; m < 4; ++m)
#pragma unroll
        for (int n = 0; n < 2; ++n)
#pragma unroll
            for (int r = 0; r < 8; ++r) {
                float v = acc[m][n][r] + bv[n];
                if (act) v = geluf_(v);
                D[(size_t)(rowW + 16*m + r + 8*hi) * ldd + colW + 16*n + rsel] = v;
            }
}

// -------------------------- prep / cast kernels ----------------------------
// src[K,N] f32 row-major -> dst[N,K] f16 row-major
__global__ void k_transpose_cast(const float* __restrict__ src,
                                 _Float16* __restrict__ dst,
                                 int K, int N, int total)
{
    int i = blockIdx.x * 256 + threadIdx.x;
    if (i >= total) return;
    int k = i / N, n = i - k * N;
    dst[(size_t)n * K + k] = (_Float16)src[i];
}

__global__ void k_cast_f16(const float* __restrict__ s, _Float16* __restrict__ d, int n)
{
    int i = blockIdx.x * 256 + threadIdx.x;
    if (i < n) d[i] = (_Float16)s[i];
}

// ------------------------------ frontend -----------------------------------
// feats (B,1600,80) -> x1 (B,800,512), stride-2 conv k=3 pad=1 + GELU
__global__ void __launch_bounds__(256)
k_conv1(const float* __restrict__ feats, const float* __restrict__ w,
        const float* __restrict__ bia, float* __restrict__ out)
{
    int o = blockIdx.x * 256 + threadIdx.x;   // 0..511
    int t = blockIdx.y;                       // 0..799
    int b = blockIdx.z;
    float acc = bia[o];
    const float* wo = w + (size_t)o * 240;    // (80,3)
#pragma unroll
    for (int k = 0; k < 3; ++k) {
        int tau = 2*t - 1 + k;
        if (tau < 0 || tau >= 1600) continue;
        const float* fr = feats + ((size_t)b*1600 + tau) * 80;
#pragma unroll 8
        for (int c = 0; c < 80; ++c) acc += fr[c] * wo[c*3 + k];
    }
    out[((size_t)b*800 + t)*512 + o] = geluf_(acc);
}

// im2col for conv2: x1 (B,800,512) -> a2 (3200, 1536) f16, K-order = c*3+k
__global__ void __launch_bounds__(256)
k_im2col2(const float* __restrict__ x1, _Float16* __restrict__ a2, int total)
{
    int idx = blockIdx.x * 256 + threadIdx.x;
    if (idx >= total) return;
    int m  = idx / 1536, ck = idx - m * 1536;
    int c  = ck / 3,  k  = ck - c * 3;
    int b  = m / 400, t  = m - b * 400;
    int tau = 2*t - 1 + k;
    float v = 0.f;
    if (tau >= 0 && tau < 800) v = x1[((size_t)b*800 + tau) * 512 + c];
    a2[idx] = (_Float16)v;
}

// ------------------------------ mamba block --------------------------------
// RMSNorm over D=512, one wave per row; emit f16 for the WMMA GEMM
__global__ void __launch_bounds__(128)
k_rmsnorm(const float* __restrict__ x, const float* __restrict__ nw,
          _Float16* __restrict__ out)
{
    int lane = threadIdx.x & 31;
    int row  = blockIdx.x * 4 + (threadIdx.x >> 5);
    const float* xr = x + (size_t)row * 512 + lane * 16;
    float v[16]; float ss = 0.f;
#pragma unroll
    for (int j = 0; j < 4; ++j) {
        float4 t = ((const float4*)xr)[j];
        v[4*j+0] = t.x; v[4*j+1] = t.y; v[4*j+2] = t.z; v[4*j+3] = t.w;
    }
#pragma unroll
    for (int j = 0; j < 16; ++j) ss += v[j]*v[j];
#pragma unroll
    for (int o = 16; o > 0; o >>= 1) ss += __shfl_xor(ss, o, 32);
    float r = rsqrtf(ss * (1.f/512.f) + 1e-5f);
    _Float16* orow = out + (size_t)row * 512 + lane * 16;
    const float* nwp = nw + lane * 16;
#pragma unroll
    for (int j = 0; j < 16; ++j) orow[j] = (_Float16)(v[j] * r * nwp[j]);
}

// causal depthwise conv (K=4) over xz[:, :1024] + SiLU -> xi (f32 + f16)
__global__ void __launch_bounds__(256)
k_dwconv(const float* __restrict__ xz, const float* __restrict__ cw,
         const float* __restrict__ cb, float* __restrict__ xi,
         _Float16* __restrict__ xi16)
{
    int idx = blockIdx.x * 256 + threadIdx.x;     // 3200*1024
    int d   = idx & 1023;
    int row = idx >> 10;
    int t   = row % 400;
    float acc = cb[d];
#pragma unroll
    for (int k = 0; k < 4; ++k) {
        int tt = t - 3 + k;
        if (tt >= 0) acc += cw[d*4 + k] * xz[(size_t)(row - 3 + k) * 2048 + d];
    }
    float s = siluf_(acc);
    xi[idx] = s;
    xi16[idx] = (_Float16)s;
}

// selective scan: thread per (b,d); 16-wide state in VGPRs; 400 sequential steps
__global__ void __launch_bounds__(256)
k_scan(const float* __restrict__ dtraw, const float* __restrict__ dt_b,
       const float* __restrict__ A_log, const float* __restrict__ proj,
       const float* __restrict__ xi, const float* __restrict__ dskip,
       float* __restrict__ ys)
{
    int d = blockIdx.x * 256 + threadIdx.x;   // 0..1023
    int b = blockIdx.y;                       // 0..7
    float Aneg[16], h[16];
#pragma unroll
    for (int n = 0; n < 16; ++n) { Aneg[n] = -__expf(A_log[(size_t)d*16 + n]); h[n] = 0.f; }
    float dbv = dt_b[d], dsk = dskip[d];
    for (int t = 0; t < 400; ++t) {
        int row = b*400 + t;
        float dtv = dtraw[(size_t)row*1024 + d] + dbv;
        dtv = (dtv > 20.f) ? dtv : log1pf(__expf(dtv));     // softplus
        float xv = xi[(size_t)row*1024 + d];
        float dx = dtv * xv;
        const float* pr = proj + (size_t)row*64;            // cols 32..47 = B, 48..63 = C
        float y = 0.f;
#pragma unroll
        for (int n = 0; n < 16; ++n) {
            float dA = __expf(dtv * Aneg[n]);
            h[n] = dA * h[n] + dx * pr[32 + n];
            y += h[n] * pr[48 + n];
        }
        ys[(size_t)row*1024 + d] = y + dsk * xv;
    }
}

// y = ys * silu(z) -> f16 for out_proj GEMM
__global__ void __launch_bounds__(256)
k_gate(const float* __restrict__ ys, const float* __restrict__ xz,
       _Float16* __restrict__ y16)
{
    int idx = blockIdx.x * 256 + threadIdx.x;     // 3200*1024
    int d = idx & 1023, row = idx >> 10;
    float z = xz[(size_t)row*2048 + 1024 + d];
    y16[idx] = (_Float16)(ys[idx] * siluf_(z));
}

__global__ void k_lens(const int* __restrict__ flens, int* __restrict__ out)
{
    int i = threadIdx.x;
    if (i < 8) { int v = flens[i] / 4; out[i] = v < 1 ? 1 : v; }
}

// ------------------------------- host --------------------------------------
extern "C" void kernel_launch(void* const* d_in, const int* in_sizes, int n_in,
                              void* d_out, int out_size, void* d_ws, size_t ws_size,
                              hipStream_t stream)
{
    (void)in_sizes; (void)n_in; (void)out_size; (void)ws_size;
    const float* feats    = (const float*)d_in[0];
    const int*   flens    = (const int*)  d_in[1];
    const float* conv1_w  = (const float*)d_in[2];
    const float* conv1_b  = (const float*)d_in[3];
    const float* conv2_w  = (const float*)d_in[4];
    const float* conv2_b  = (const float*)d_in[5];
    const float* norm_w   = (const float*)d_in[6];
    const float* in_proj  = (const float*)d_in[7];
    const float* dw_w     = (const float*)d_in[8];
    const float* dw_b     = (const float*)d_in[9];
    const float* x_proj   = (const float*)d_in[10];
    const float* dt_w     = (const float*)d_in[11];
    const float* dt_b     = (const float*)d_in[12];
    const float* A_log    = (const float*)d_in[13];
    const float* D_skip   = (const float*)d_in[14];
    const float* out_proj = (const float*)d_in[15];
    const float* head_w   = (const float*)d_in[16];
    const float* head_b   = (const float*)d_in[17];

    const int M = 3200;               // B*T4
    char* ws = (char*)d_ws;
    size_t off = 0;
    auto alloc = [&](size_t bytes) -> char* {
        off = (off + 255) & ~(size_t)255;
        char* p = ws + off; off += bytes; return p;
    };

    _Float16* w_in16   = (_Float16*)alloc((size_t)8*2048*512*2);
    _Float16* w_xp16   = (_Float16*)alloc((size_t)8*64*1024*2);
    _Float16* w_dt16   = (_Float16*)alloc((size_t)8*1024*32*2);
    _Float16* w_out16  = (_Float16*)alloc((size_t)8*512*1024*2);
    _Float16* w_head16 = (_Float16*)alloc((size_t)1024*512*2);
    _Float16* w2_16    = (_Float16*)alloc((size_t)512*1536*2);
    float*    x1       = (float*)   alloc((size_t)8*800*512*4);
    _Float16* a2       = (_Float16*)alloc((size_t)M*1536*2);
    float*    x        = (float*)   alloc((size_t)M*512*4);
    _Float16* xn16     = (_Float16*)alloc((size_t)M*512*2);
    float*    xz       = (float*)   alloc((size_t)M*2048*4);
    float*    xi       = (float*)   alloc((size_t)M*1024*4);
    _Float16* xi16     = (_Float16*)alloc((size_t)M*1024*2);
    float*    proj     = (float*)   alloc((size_t)M*64*4);
    _Float16* proj16   = (_Float16*)alloc((size_t)M*64*2);
    float*    dtraw    = (float*)   alloc((size_t)M*1024*4);
    float*    ys       = (float*)   alloc((size_t)M*1024*4);
    _Float16* y16      = (_Float16*)alloc((size_t)M*1024*2);
    _Float16* x16      = (_Float16*)alloc((size_t)M*512*2);

    // ---- weight prep ----
    for (int l = 0; l < 8; ++l) {
        { int K=512,  N=2048, tot=K*N;
          k_transpose_cast<<<(tot+255)/256,256,0,stream>>>(in_proj  + (size_t)l*tot, w_in16  + (size_t)l*tot, K, N, tot); }
        { int K=1024, N=64,   tot=K*N;
          k_transpose_cast<<<(tot+255)/256,256,0,stream>>>(x_proj   + (size_t)l*tot, w_xp16  + (size_t)l*tot, K, N, tot); }
        { int K=32,   N=1024, tot=K*N;
          k_transpose_cast<<<(tot+255)/256,256,0,stream>>>(dt_w     + (size_t)l*tot, w_dt16  + (size_t)l*tot, K, N, tot); }
        { int K=1024, N=512,  tot=K*N;
          k_transpose_cast<<<(tot+255)/256,256,0,stream>>>(out_proj + (size_t)l*tot, w_out16 + (size_t)l*tot, K, N, tot); }
    }
    { int K=512, N=1024, tot=K*N;
      k_transpose_cast<<<(tot+255)/256,256,0,stream>>>(head_w, w_head16, K, N, tot); }
    // conv2_w (512,512,3) is already N x K with K-order c*3+k: cast only
    k_cast_f16<<<(512*1536+255)/256, 256, 0, stream>>>(conv2_w, w2_16, 512*1536);

    // ---- frontend ----
    k_conv1<<<dim3(2, 800, 8), 256, 0, stream>>>(feats, conv1_w, conv1_b, x1);
    k_im2col2<<<(M*1536+255)/256, 256, 0, stream>>>(x1, a2, M*1536);
    // x = gelu(im2col(x1) @ conv2_w^T + conv2_b) : (3200,1536)x(1536,512)
    k_wmma_gemm<<<dim3(512/64, M/128), 128, 0, stream>>>(
        a2, 1536, w2_16, 1536, nullptr, conv2_b, x, 512, 1536, 1);

    // ---- 8 mamba blocks ----
    const int nElemDI = M * 1024;
    for (int l = 0; l < 8; ++l) {
        k_rmsnorm<<<M/4, 128, 0, stream>>>(x, norm_w + (size_t)l*512, xn16);

        // xz = xn @ in_proj : (3200,512)x(512,2048)
        k_wmma_gemm<<<dim3(2048/64, M/128), 128, 0, stream>>>(
            xn16, 512, w_in16 + (size_t)l*2048*512, 512,
            nullptr, nullptr, xz, 2048, 512, 0);

        k_dwconv<<<(nElemDI+255)/256, 256, 0, stream>>>(
            xz, dw_w + (size_t)l*1024*4, dw_b + (size_t)l*1024, xi, xi16);

        // proj = xi @ x_proj : (3200,1024)x(1024,64)
        k_wmma_gemm<<<dim3(64/64, M/128), 128, 0, stream>>>(
            xi16, 1024, w_xp16 + (size_t)l*64*1024, 1024,
            nullptr, nullptr, proj, 64, 1024, 0);

        k_cast_f16<<<(M*64+255)/256, 256, 0, stream>>>(proj, proj16, M*64);

        // dtraw = proj[:, :32] @ dt_w : (3200,32)x(32,1024)
        k_wmma_gemm<<<dim3(1024/64, M/128), 128, 0, stream>>>(
            proj16, 64, w_dt16 + (size_t)l*1024*32, 32,
            nullptr, nullptr, dtraw, 1024, 32, 0);

        k_scan<<<dim3(4, 8), 256, 0, stream>>>(
            dtraw, dt_b + (size_t)l*1024, A_log + (size_t)l*1024*16,
            proj, xi, D_skip + (size_t)l*1024, ys);

        k_gate<<<(nElemDI+255)/256, 256, 0, stream>>>(ys, xz, y16);

        // x += y @ out_proj : fused residual via Cin=x (in place, disjoint per lane)
        k_wmma_gemm<<<dim3(512/64, M/128), 128, 0, stream>>>(
            y16, 1024, w_out16 + (size_t)l*512*1024, 1024,
            x, nullptr, x, 512, 1024, 0);
    }

    // ---- head: logits = x @ head_w + head_b -> d_out ----
    k_cast_f16<<<(M*512+255)/256, 256, 0, stream>>>(x, x16, M*512);
    k_wmma_gemm<<<dim3(1024/64, M/128), 128, 0, stream>>>(
        x16, 512, w_head16, 512,
        nullptr, head_b, (float*)d_out, 1024, 512, 0);

    // ---- out_lens ----
    k_lens<<<1, 32, 0, stream>>>(flens, (int*)d_out + (size_t)M*1024);
}
